// kernel_generated_0_62904091017264
// MI455X (gfx1250) — matrix operand order repaired and checked by reading, not compiled, not run on hardware
//
#include <hip/hip_runtime.h>

// ---------------------------------------------------------------------------
// Shapes (fixed by the reference)
#define B_   1024
#define C_   64
#define H_   56
#define W_   56
#define PQS  32                     // bf16 elems per (row,tt) line (64 B)
#define ROWS 4                      // staged h rows per WG -> 2 output rows
#define PQ_ELEMS (ROWS * 58 * PQS)  // 7424 bf16 per array
#define W_ELEMS  (9 * 64 * PQS)     // 18432 bf16
#define SMEM_BYTES ((2 * PQ_ELEMS + W_ELEMS) * 2)  // 66560 B (dynamic LDS)

typedef __attribute__((ext_vector_type(4)))  float  v4f;
typedef __attribute__((ext_vector_type(8)))  float  v8f;
typedef __attribute__((ext_vector_type(8)))  __bf16 v8bf;
typedef __attribute__((ext_vector_type(16))) __bf16 v16bf;

union U16B { uint4 u; v8bf b; };

__device__ __forceinline__ v16bf cat16(v8bf lo, v8bf hi) {
  return __builtin_shufflevector(lo, hi, 0,1,2,3,4,5,6,7,8,9,10,11,12,13,14,15);
}

__device__ __forceinline__ v8f wmma_bf16(v16bf a, v16bf b, v8f c) {
  return __builtin_amdgcn_wmma_f32_16x16x32_bf16(false, a, false, b,
                                                 (short)0, c, false, false);
}

// ---------------------------------------------------------------------------
// Prep: weights f32 (l,j,k,i) -> bf16 in B-matrix layout [tap=k*3+j][i][l]
__global__ void prep_weights_bf16(const float* __restrict__ w,
                                  __bf16* __restrict__ wbf) {
  int e = blockIdx.x * 256 + threadIdx.x;     // over 9*64*32
  if (e >= 9 * 64 * 32) return;
  int l   = e & 31;
  int i   = (e >> 5) & 63;
  int tap = e >> 11;                          // 0..8 == k*3 + j
  int k = tap / 3, j = tap - 3 * k;
  float v = w[((l * 3 + j) * 3 + k) * 64 + i];
  wbf[(tap * 64 + i) * PQS + l] = (__bf16)v;
}

// ---------------------------------------------------------------------------
// Main: one WG per (b, h-pair). M=64 (w; 56 valid) x N=64 (i), K=32 (l) x 9
// taps, two output rows h0,h0+1 sharing the staged x rows h0-1..h0+2.
__global__ __launch_bounds__(256)
void shiftconv_wmma2(const float* __restrict__ x,
                     const __bf16* __restrict__ wbf,
                     float* __restrict__ out) {
  extern __shared__ __align__(16) char smem[];
  __bf16* Plds = (__bf16*)smem;            // [jj<4][tt<58][l<32] channels 0..31
  __bf16* Qlds = Plds + PQ_ELEMS;          // same, channels 32..63
  __bf16* Wlds = Qlds + PQ_ELEMS;          // [tap<9][i<64][l<32]

  const int tid = threadIdx.x;
  const int blk = blockIdx.x;
  const int b   = blk / (H_ / 2);
  const int h0  = (blk - b * (H_ / 2)) * 2;

  // ---- stage weights (contiguous copy, layout already final) ----
  {
    const uint4* src = (const uint4*)wbf;
    uint4*       dst = (uint4*)Wlds;
    for (int idx = tid; idx < (W_ELEMS * 2) / 16; idx += 256)
      dst[idx] = src[idx];
  }

  // ---- stage P/Q: coalesced float4 row loads, scatter-store bf16 ----
  // P[jj][tt][l] = x[b, l, h0+jj-1, (tt-2+56)%56]  (roll +1 baked into tt;
  // tt=0 / tt=57 are the zero-pad rows written below)
  for (int e = tid; e < C_ * ROWS * 14; e += 256) {
    int c   = e / (ROWS * 14);        // channel 0..63
    int rem = e - c * (ROWS * 14);
    int jj  = rem / 14;               // staged row 0..3
    int q   = rem - jj * 14;          // float4 index along w
    int hh  = h0 + jj - 1;
    float4 v = make_float4(0.f, 0.f, 0.f, 0.f);
    if (hh >= 0 && hh < H_)
      v = *(const float4*)(x + (((size_t)b * C_ + c) * H_ + hh) * W_ + q * 4);
    __bf16* base = (c < 32 ? Plds : Qlds) + (jj * 58) * PQS + (c & 31);
    float vv[4] = {v.x, v.y, v.z, v.w};
#pragma unroll
    for (int u = 0; u < 4; ++u) {
      int ws = q * 4 + u;
      int tt = (ws == 55) ? 1 : ws + 2;       // circular roll(+1)
      base[tt * PQS] = (__bf16)vv[u];
    }
  }
  // zero-pad rows tt=0 and tt=57 (2 arrays x 4 rows x 2 tt x 32 l)
  for (int e = tid; e < 512; e += 256) {
    int l  = e & 31;
    int r2 = e >> 5;                  // 0..15
    int jj = r2 & 3;
    int r3 = r2 >> 2;                 // 0..3
    __bf16* arr = (r3 & 1) ? Qlds : Plds;
    int tt = (r3 >> 1) ? 57 : 0;
    arr[(jj * 58 + tt) * PQS + l] = (__bf16)0.0f;
  }
  __syncthreads();

  // ---- WMMA main loop ----
  const int wv    = tid >> 5;            // wave 0..7
  const int lane  = tid & 31;
  const int lrow  = lane & 15;
  const int lhalf = lane >> 4;
  const int mtile = wv & 3;              // w-tile
  const int nb    = (wv >> 2) * 32;      // channel base (two 16-wide N tiles)
  const int w0    = mtile * 16 + lrow;   // A row (w index)
  const int wm1   = (w0 + 55) % 56;      // exact circular-roll index for tb
  const int c0    = lhalf * 8;           // ISA 16-bit A layout K-chunks
  const int c1    = 16 + lhalf * 8;

  v8f acc[2][2] = {};                    // [output row r][N tile]

#pragma unroll
  for (int k = 0; k < 3; ++k) {
    // preload B fragments for taps (k, j=0..2), both N tiles
    v16bf Bf[3][2];
#pragma unroll
    for (int j = 0; j < 3; ++j) {
      const __bf16* wc = Wlds + ((k * 3 + j) * 64 + nb + lrow) * PQS + lhalf * 8;
      U16B b0, b1, b2, b3;
      b0.u = *(const uint4*)(wc);
      b1.u = *(const uint4*)(wc + 16);
      b2.u = *(const uint4*)(wc + 16 * PQS);
      b3.u = *(const uint4*)(wc + 16 * PQS + 16);
      Bf[j][0] = cat16(b0.b, b1.b);
      Bf[j][1] = cat16(b2.b, b3.b);
    }
    int tt1 = w0 + k;  if (tt1 > 57) tt1 = 57;   // clamp only for padded w>=56
    const int tt2 = wm1 + k;                      // always in [0,57]
#pragma unroll
    for (int jj = 0; jj < ROWS; ++jj) {
      const __bf16* prow = Plds + (jj * 58 + tt1) * PQS;
      const __bf16* qrow = Qlds + (jj * 58 + tt2) * PQS;
      U16B p0, p1, q0, q1;
      p0.u = *(const uint4*)(prow + c0);
      p1.u = *(const uint4*)(prow + c1);
      q0.u = *(const uint4*)(qrow + c0);
      q1.u = *(const uint4*)(qrow + c1);
      v16bf af = cat16(p0.b + q0.b, p1.b + q1.b);  // ta + tb (packed bf16)
      if (jj < 3) {            // output row r=0 uses tap j = jj
        acc[0][0] = wmma_bf16(af, Bf[jj][0], acc[0][0]);
        acc[0][1] = wmma_bf16(af, Bf[jj][1], acc[0][1]);
      }
      if (jj >= 1) {           // output row r=1 uses tap j = jj-1
        acc[1][0] = wmma_bf16(af, Bf[jj - 1][0], acc[1][0]);
        acc[1][1] = wmma_bf16(af, Bf[jj - 1][1], acc[1][1]);
      }
    }
  }

  // ---- store C: lane holds N=lrow, M = r + 8*lhalf -> contiguous w runs ----
  const int wb = mtile * 16 + lhalf * 8;
  if (wb < 56) {
    const int i0 = nb + lrow;
#pragma unroll
    for (int r = 0; r < 2; ++r) {
      const int hcur = h0 + r;
      const v4f* a0 = (const v4f*)&acc[r][0];
      const v4f* a1 = (const v4f*)&acc[r][1];
      v4f* d0 = (v4f*)(out + (((size_t)b * C_ + i0) * H_ + hcur) * W_ + wb);
      __builtin_nontemporal_store(a0[0], d0);
      __builtin_nontemporal_store(a0[1], d0 + 1);
      v4f* d1 = (v4f*)(out + (((size_t)b * C_ + i0 + 16) * H_ + hcur) * W_ + wb);
      __builtin_nontemporal_store(a1[0], d1);
      __builtin_nontemporal_store(a1[1], d1 + 1);
    }
  }
}

// ---------------------------------------------------------------------------
extern "C" void kernel_launch(void* const* d_in, const int* in_sizes, int n_in,
                              void* d_out, int out_size, void* d_ws, size_t ws_size,
                              hipStream_t stream) {
  const float* x = (const float*)d_in[0];   // (1024, 64, 56, 56) f32
  const float* w = (const float*)d_in[1];   // (32, 3, 3, 64)     f32
  float* out = (float*)d_out;               // (1024, 64, 56, 56) f32
  __bf16* wbf = (__bf16*)d_ws;              // 9*64*32 bf16 = 36864 B scratch

  prep_weights_bf16<<<(9 * 64 * 32 + 255) / 256, 256, 0, stream>>>(w, wbf);
  shiftconv_wmma2<<<B_ * (H_ / 2), 256, SMEM_BYTES, stream>>>(x, wbf, out);
}
